// CoT_8933531976327
// MI455X (gfx1250) — matrix operand order repaired and checked by reading, not compiled, not run on hardware
//
#include <hip/hip_runtime.h>
#include <math.h>

// ---------------- problem constants (from reference) ----------------
#define B_GRAPHS 32
#define NPG      2048
#define EPG      16384
#define NN       (B_GRAPHS * NPG)   // 65536 nodes
#define EE       (B_GRAPHS * EPG)   // 524288 edges
#define K_SEL    8192

typedef _Float16 half_t;
typedef __attribute__((ext_vector_type(16))) _Float16 v16h;
typedef __attribute__((ext_vector_type(8)))  _Float16 v8h;
typedef __attribute__((ext_vector_type(4)))  _Float16 v4h;
typedef __attribute__((ext_vector_type(8)))  float    v8f;
typedef __attribute__((ext_vector_type(4)))  float    v4f;

#define NWAVES 4
#define BLOCK  (NWAVES * 32)
#define GRID   1024
#define NTILES (EE / 16)            // 32768 tiles of 16 edges

// ---------------- LDS layout ----------------
struct Weights {
  half_t ftWt[64 * 128];   // ft_W rows 1..128, transposed [n][k], f16
  half_t m0Wt[64 * 128];   // m0_W rows 1..128, transposed [n][k], f16
  half_t m1Wt[32 * 64];    // m1_W transposed [n][k], f16
  float  ftW0[64], ftb[64];    // ft_W row 0 (degree term) + bias, f32
  float  m0W0[64], m0b[64];    // m0_W row 0 (sim term) + bias, f32
  float  m1b[32], m2w[32], m2b, pad[3];
};

struct alignas(64) WaveScratch {
  half_t Atile[16 * 128];  // 16-edge x 128-K f16 A-tile (h gather / [rf,cf])
  half_t A2[16 * 64];      // relu(m0) f16 A-tile for m1
  int    eRow[16], eCol[16];
  float  eDegR[16], eDegC[16];
  float  logit[16];
  float  padw[4];
};

struct SmemLayout { Weights w; WaveScratch wv[NWAVES]; };  // ~63 KB

extern __shared__ __align__(64) char smem_raw[];

// ---------------- fragment loaders (wave32 WMMA layouts) ----------------
// 16-bit A 16x32: lanes 0-15 -> K 0..15, lanes 16-31 -> K 16..31 (matches B)
__device__ __forceinline__ v16h load_a_frag(const half_t* At, int strideH,
                                            int lane, int kc) {
  const int m  = lane & 15;
  const int kb = kc * 32 + ((lane & 16) ? 16 : 0);
  const half_t* p = At + m * strideH + kb;
  v8h lo = *(const v8h*)p;
  v8h hh = *(const v8h*)(p + 8);
  v16h a;
#pragma unroll
  for (int t = 0; t < 8; ++t) { a[t] = lo[t]; a[t + 8] = hh[t]; }
  return a;
}
// 16-bit B 32x16: lane holds column n = lane%16, K = 16 contiguous values
// (lanes 0-15: K 0..15; lanes 16-31: K 16..31).  Weights stored transposed
// [n][k] in LDS -> one contiguous 32B v16h load.
__device__ __forceinline__ v16h load_b_frag(const half_t* Wt, int strideH,
                                            int lane, int nt, int kc) {
  const int n  = nt * 16 + (lane & 15);
  const int kb = kc * 32 + ((lane & 16) ? 16 : 0);
  return *(const v16h*)(Wt + n * strideH + kb);
}

// sum across a 16-lane half-wave (xor masks < 16 stay within the half)
__device__ __forceinline__ float red16(float x) {
  x += __shfl_xor(x, 1, 32);
  x += __shfl_xor(x, 2, 32);
  x += __shfl_xor(x, 4, 32);
  x += __shfl_xor(x, 8, 32);
  return x;
}

// ---------------- degree kernels ----------------
__global__ void zero_deg_kernel(float* deg) {
  int i = blockIdx.x * blockDim.x + threadIdx.x;
  if (i < NN) deg[i] = 0.f;
}
__global__ void degree_kernel(const int* __restrict__ row,
                              const int* __restrict__ col,
                              float* __restrict__ deg) {
  int e = blockIdx.x * blockDim.x + threadIdx.x;
  if (e < EE) {
    atomicAdd(&deg[row[e]], 1.f);
    atomicAdd(&deg[col[e]], 1.f);
  }
}

// ---------------- main edge-score kernel ----------------
__global__ __launch_bounds__(BLOCK) void edge_score_kernel(
    const float* __restrict__ h, const int* __restrict__ row,
    const int* __restrict__ col,
    const float* __restrict__ ftW, const float* __restrict__ ftb_g,
    const float* __restrict__ m0W, const float* __restrict__ m0b_g,
    const float* __restrict__ m1W, const float* __restrict__ m1b_g,
    const float* __restrict__ m2W, const float* __restrict__ m2b_g,
    const float* __restrict__ deg, float* __restrict__ scores) {
  SmemLayout& S = *reinterpret_cast<SmemLayout*>(smem_raw);
  const int tid = threadIdx.x;

  // ---- cooperative weight preload: f32 -> f16, transposed [n][k] ----
  for (int i = tid; i < 64 * 128; i += BLOCK) {
    const int n = i >> 7, k = i & 127;
    S.w.ftWt[i] = (half_t)ftW[(k + 1) * 64 + n];
    S.w.m0Wt[i] = (half_t)m0W[(k + 1) * 64 + n];
  }
  for (int i = tid; i < 32 * 64; i += BLOCK) {
    const int n = i >> 6, k = i & 63;
    S.w.m1Wt[i] = (half_t)m1W[k * 32 + n];
  }
  if (tid < 64) {
    S.w.ftW0[tid] = ftW[tid];   S.w.ftb[tid] = ftb_g[tid];
    S.w.m0W0[tid] = m0W[tid];   S.w.m0b[tid] = m0b_g[tid];
  }
  if (tid < 32) { S.w.m1b[tid] = m1b_g[tid]; S.w.m2w[tid] = m2W[tid]; }
  if (tid == 0) S.w.m2b = m2b_g[0];
  __syncthreads();

  const int lane = tid & 31;
  const int wid  = tid >> 5;
  const int hg   = lane >> 4;        // half-wave group (0/1) -> row offset 8*hg
  const int nlo  = lane & 15;        // D-fragment column within tile
  WaveScratch& W = S.wv[wid];

  const int waveGlobal = blockIdx.x * NWAVES + wid;
  const int waveStride = gridDim.x * NWAVES;

  for (int tile = waveGlobal; tile < NTILES; tile += waveStride) {
    const int eBase = tile * 16;

    // ---- edge metadata + degree gather ----
    if (lane < 16) {
      const int r = row[eBase + lane];
      const int c = col[eBase + lane];
      W.eRow[lane] = r;      W.eCol[lane] = c;
      W.eDegR[lane] = deg[r]; W.eDegC[lane] = deg[c];
    }
    __builtin_amdgcn_wave_barrier();

    v8f rfD[4], cfD[4];

    // ---- gather h[row] -> f16 A-tile ----
#pragma unroll
    for (int it = 0; it < 16; ++it) {
      const int i = it * 32 + lane;          // 512 float4 slots
      const int el = i >> 5, q = i & 31;
      const v4f src = *(const v4f*)(h + (size_t)W.eRow[el] * 128 + q * 4);
      v4h d;
#pragma unroll
      for (int t = 0; t < 4; ++t) d[t] = (half_t)src[t];
      *(v4h*)(W.Atile + el * 128 + q * 4) = d;
    }
    __builtin_amdgcn_wave_barrier();

    // rf: C init = bias + deg * ft_W[0,:]  (rank-1 fold of the 129th K dim)
#pragma unroll
    for (int nt = 0; nt < 4; ++nt) {
      const int cg = nt * 16 + nlo;
      const float bias = S.w.ftb[cg], w0 = S.w.ftW0[cg];
#pragma unroll
      for (int v = 0; v < 8; ++v) rfD[nt][v] = bias + W.eDegR[v + 8 * hg] * w0;
    }
#pragma unroll
    for (int kc = 0; kc < 4; ++kc) {
      const v16h a = load_a_frag(W.Atile, 128, lane, kc);
#pragma unroll
      for (int nt = 0; nt < 4; ++nt) {
        const v16h b = load_b_frag(S.w.ftWt, 128, lane, nt, kc);
        rfD[nt] = __builtin_amdgcn_wmma_f32_16x16x32_f16(
            false, a, false, b, (short)0, rfD[nt], false, false);
      }
    }

    // ---- gather h[col] -> A-tile, compute cf ----
    __builtin_amdgcn_wave_barrier();
#pragma unroll
    for (int it = 0; it < 16; ++it) {
      const int i = it * 32 + lane;
      const int el = i >> 5, q = i & 31;
      const v4f src = *(const v4f*)(h + (size_t)W.eCol[el] * 128 + q * 4);
      v4h d;
#pragma unroll
      for (int t = 0; t < 4; ++t) d[t] = (half_t)src[t];
      *(v4h*)(W.Atile + el * 128 + q * 4) = d;
    }
    __builtin_amdgcn_wave_barrier();
#pragma unroll
    for (int nt = 0; nt < 4; ++nt) {
      const int cg = nt * 16 + nlo;
      const float bias = S.w.ftb[cg], w0 = S.w.ftW0[cg];
#pragma unroll
      for (int v = 0; v < 8; ++v) cfD[nt][v] = bias + W.eDegC[v + 8 * hg] * w0;
    }
#pragma unroll
    for (int kc = 0; kc < 4; ++kc) {
      const v16h a = load_a_frag(W.Atile, 128, lane, kc);
#pragma unroll
      for (int nt = 0; nt < 4; ++nt) {
        const v16h b = load_b_frag(S.w.ftWt, 128, lane, nt, kc);
        cfD[nt] = __builtin_amdgcn_wmma_f32_16x16x32_f16(
            false, a, false, b, (short)0, cfD[nt], false, false);
      }
    }

    // ---- cosine similarity per edge (row-wise reduce over 64 cols) ----
    // D layout: lane=col (nlo), VGPR v = row v+8*hg.  Reduced sums land in
    // exactly the lane/VGPR slot the m0 accumulator init needs.
    float simv[8];
#pragma unroll
    for (int v = 0; v < 8; ++v) {
      float pd = 0.f, pr = 0.f, pc = 0.f;
#pragma unroll
      for (int nt = 0; nt < 4; ++nt) {
        const float a = rfD[nt][v], b = cfD[nt][v];
        pd += a * b; pr += a * a; pc += b * b;
      }
      pd = red16(pd); pr = red16(pr); pc = red16(pc);
      const float den = sqrtf(pr) * sqrtf(pc);
      simv[v] = pd / fmaxf(den, 1e-8f);
    }

    // ---- pack [rf, cf] as f16 A-tile for the MLP ----
    __builtin_amdgcn_wave_barrier();
#pragma unroll
    for (int nt = 0; nt < 4; ++nt) {
      const int cg = nt * 16 + nlo;
#pragma unroll
      for (int v = 0; v < 8; ++v) {
        const int r = v + 8 * hg;
        W.Atile[r * 128 + cg]      = (half_t)rfD[nt][v];
        W.Atile[r * 128 + 64 + cg] = (half_t)cfD[nt][v];
      }
    }
    __builtin_amdgcn_wave_barrier();

    // ---- m0: C init = bias + sim * m0_W[0,:]  (rank-1 fold of sim) ----
    v8f m0D[4];
#pragma unroll
    for (int nt = 0; nt < 4; ++nt) {
      const int cg = nt * 16 + nlo;
      const float bias = S.w.m0b[cg], w0 = S.w.m0W0[cg];
#pragma unroll
      for (int v = 0; v < 8; ++v) m0D[nt][v] = bias + simv[v] * w0;
    }
#pragma unroll
    for (int kc = 0; kc < 4; ++kc) {
      const v16h a = load_a_frag(W.Atile, 128, lane, kc);
#pragma unroll
      for (int nt = 0; nt < 4; ++nt) {
        const v16h b = load_b_frag(S.w.m0Wt, 128, lane, nt, kc);
        m0D[nt] = __builtin_amdgcn_wmma_f32_16x16x32_f16(
            false, a, false, b, (short)0, m0D[nt], false, false);
      }
    }
    // relu -> A2
#pragma unroll
    for (int nt = 0; nt < 4; ++nt) {
      const int cg = nt * 16 + nlo;
#pragma unroll
      for (int v = 0; v < 8; ++v)
        W.A2[(v + 8 * hg) * 64 + cg] = (half_t)fmaxf(m0D[nt][v], 0.f);
    }
    __builtin_amdgcn_wave_barrier();

    // ---- m1: 64 -> 32 ----
    v8f m1D[2];
#pragma unroll
    for (int nt = 0; nt < 2; ++nt) {
      const float bias = S.w.m1b[nt * 16 + nlo];
#pragma unroll
      for (int v = 0; v < 8; ++v) m1D[nt][v] = bias;
    }
#pragma unroll
    for (int kc = 0; kc < 2; ++kc) {
      const v16h a = load_a_frag(W.A2, 64, lane, kc);
#pragma unroll
      for (int nt = 0; nt < 2; ++nt) {
        const v16h b = load_b_frag(S.w.m1Wt, 64, lane, nt, kc);
        m1D[nt] = __builtin_amdgcn_wmma_f32_16x16x32_f16(
            false, a, false, b, (short)0, m1D[nt], false, false);
      }
    }

    // ---- m2 (32 -> 1) + sigmoid ----
#pragma unroll
    for (int v = 0; v < 8; ++v) {
      float p = 0.f;
#pragma unroll
      for (int nt = 0; nt < 2; ++nt)
        p += fmaxf(m1D[nt][v], 0.f) * S.w.m2w[nt * 16 + nlo];
      p = red16(p);
      if (nlo == 0) W.logit[8 * hg + v] = p + S.w.m2b;
    }
    __builtin_amdgcn_wave_barrier();
    if (lane < 16) {
      const float x = W.logit[lane];
      scores[eBase + lane] = 1.f / (1.f + expf(-x));
    }
    __builtin_amdgcn_wave_barrier();
  }
}

// ---------------- per-graph top-K via full bitonic sort in LDS ----------------
__global__ __launch_bounds__(1024) void topk_kernel(
    const float* __restrict__ scores, const int* __restrict__ largestPtr,
    float* __restrict__ mask_out, int* __restrict__ idx_out) {
  extern __shared__ __align__(64) char sm[];
  float* v = (float*)sm;
  int*   ix = (int*)(sm + EPG * sizeof(float));
  const int b = blockIdx.x;
  const int largest = *largestPtr;

  for (int i = threadIdx.x; i < EPG; i += blockDim.x) {
    v[i] = scores[b * EPG + i];
    ix[i] = i;
  }
  __syncthreads();

  for (int k = 2; k <= EPG; k <<= 1) {
    for (int j = k >> 1; j > 0; j >>= 1) {
      for (int i = threadIdx.x; i < EPG; i += blockDim.x) {
        const int p = i ^ j;
        if (p > i) {
          bool descend = ((i & k) == 0);
          if (!largest) descend = !descend;
          const float a0 = v[i], a1 = v[p];
          const bool sw = descend ? (a0 < a1) : (a0 > a1);
          if (sw) {
            v[i] = a1; v[p] = a0;
            const int t = ix[i]; ix[i] = ix[p]; ix[p] = t;
          }
        }
      }
      __syncthreads();
    }
  }

  for (int i = threadIdx.x; i < K_SEL; i += blockDim.x) {
    mask_out[b * K_SEL + i] = v[i];
    idx_out[b * K_SEL + i]  = ix[i] + b * EPG;
  }
}

// ---------------- launch ----------------
extern "C" void kernel_launch(void* const* d_in, const int* in_sizes, int n_in,
                              void* d_out, int out_size, void* d_ws,
                              size_t ws_size, hipStream_t stream) {
  const float* h    = (const float*)d_in[0];
  const int*   row  = (const int*)d_in[1];
  const int*   col  = (const int*)d_in[2];
  const float* ftW  = (const float*)d_in[3];
  const float* ftb  = (const float*)d_in[4];
  const float* m0W  = (const float*)d_in[5];
  const float* m0b  = (const float*)d_in[6];
  const float* m1W  = (const float*)d_in[7];
  const float* m1b  = (const float*)d_in[8];
  const float* m2W  = (const float*)d_in[9];
  const float* m2b  = (const float*)d_in[10];
  const int*   largest = (const int*)d_in[11];

  // d_out layout: edge_mask [B*K_SEL] f32 | scores [E] f32 | edge_idx [B*K_SEL] i32
  float* mask_out = (float*)d_out;
  float* scores   = mask_out + B_GRAPHS * K_SEL;
  int*   idx_out  = (int*)(mask_out + B_GRAPHS * K_SEL + EE);
  float* deg      = (float*)d_ws;   // NN floats of scratch

  zero_deg_kernel<<<(NN + 255) / 256, 256, 0, stream>>>(deg);
  degree_kernel<<<(EE + 255) / 256, 256, 0, stream>>>(row, col, deg);

  const size_t smem = sizeof(SmemLayout);
  edge_score_kernel<<<GRID, BLOCK, smem, stream>>>(
      h, row, col, ftW, ftb, m0W, m0b, m1W, m1b, m2W, m2b, deg, scores);

  const size_t smem2 = (size_t)EPG * (sizeof(float) + sizeof(int));
  topk_kernel<<<B_GRAPHS, 1024, smem2, stream>>>(scores, largest, mask_out,
                                                 idx_out);
}
